// SigMMD_naive_42984032698501
// MI455X (gfx1250) — compile-verified, matrix operand order checked by reading, not run on hardware
//
#include <hip/hip_runtime.h>
#include <hip/hip_bf16.h>

typedef _Float16 v16h __attribute__((ext_vector_type(16)));
typedef _Float16 v8h  __attribute__((ext_vector_type(8)));
typedef float    v8f  __attribute__((ext_vector_type(8)));

#define AB   16
#define TLEN 256          // 255 valid increment rows + 1 zero-pad row
#define DIM  64
#define NPTS (AB * TLEN * DIM)

// ---------------------------------------------------------------- zero output
__global__ void sig_zero_out(float* out) { out[0] = 0.0f; }

// ------------------------------------------------ deltas -> f16 hi/lo + MSE(0)
__global__ void sig_prep(const float* __restrict__ X, const float* __restrict__ Y,
                         _Float16* __restrict__ dXh, _Float16* __restrict__ dXl,
                         _Float16* __restrict__ dYh, _Float16* __restrict__ dYl,
                         float* __restrict__ out)
{
    int idx = blockIdx.x * blockDim.x + threadIdx.x;
    if (idx >= NPTS) return;
    int i = (idx / DIM) & (TLEN - 1);

    float dx = 0.0f, dy = 0.0f;
    if (i < TLEN - 1) {                 // row 255 stays zero (padding)
        dx = X[idx + DIM] - X[idx];
        dy = Y[idx + DIM] - Y[idx];
    }
    _Float16 hx = (_Float16)dx;
    dXh[idx] = hx; dXl[idx] = (_Float16)(dx - (float)hx);
    _Float16 hy = (_Float16)dy;
    dYh[idx] = hy; dYl[idx] = (_Float16)(dy - (float)hy);

    if (i == 0) {                       // mean((X[:,0,:]-Y[:,0,:])^2) term
        float diff = X[idx] - Y[idx];
        atomicAdd(out, diff * diff * (1.0f / (AB * DIM)));
    }
}

// Load one 16-half WMMA fragment: two contiguous 16-byte chunks per lane,
// matching the CDNA5 16-bit A/B VGPR layout (K chunks {0..7} and {16..23}
// relative to the lane's K base).
__device__ __forceinline__ v16h sig_load_frag(const _Float16* p) {
    v8h c0 = *(const v8h*)(p);
    v8h c1 = *(const v8h*)(p + 16);
    v16h f;
#pragma unroll
    for (int k = 0; k < 8; ++k) { f[k] = c0[k]; f[8 + k] = c1[k]; }
    return f;
}

// ------------------------- fused: WMMA increments + prefix-scan PDE per pair
__global__ __launch_bounds__(256) void sig_pde(
    const _Float16* __restrict__ dXh, const _Float16* __restrict__ dXl,
    const _Float16* __restrict__ dYh, const _Float16* __restrict__ dYl,
    float* __restrict__ out)
{
    __shared__ float incS[16][257];     // 16-row inc slab (col 255 unused)
    __shared__ float Krow[256];         // current PDE row K[i, 0..255]
    __shared__ float waveSum[8];

    const int tid  = threadIdx.x;
    const int lane = tid & 31;
    const int wave = tid >> 5;

    const int mat  = blockIdx.x >> 8;   // 0: XX, 1: YY, 2: XY
    const int pair = blockIdx.x & 255;
    const int a = pair >> 4, b = pair & 15;

    const _Float16* Ph = (mat == 1) ? dYh : dXh;
    const _Float16* Pl = (mat == 1) ? dYl : dXl;
    const _Float16* Qh = (mat == 0) ? dXh : dYh;
    const _Float16* Ql = (mat == 0) ? dXl : dYl;

    const size_t arow = ((size_t)a * TLEN + (lane & 15)) * DIM;
    const size_t brow = ((size_t)b * TLEN + (lane & 15)) * DIM;
    const int dsel = (lane >> 4) * 8;   // lanes 16..31 take the K=8.. half

    Krow[tid] = 1.0f;                   // K[0, :] = 1
    __syncthreads();

    for (int i0 = 0; i0 < 255; i0 += 16) {
        // ---- Phase 1: inc rows i0..i0+15 x cols 0..255 via hi/lo f16 WMMA
#pragma unroll
        for (int tt = 0; tt < 2; ++tt) {
            const int jt = wave * 2 + tt;            // j-tile 0..15
            v8f acc = {};
            const _Float16* Ah_p = Ph + arow + (size_t)i0 * DIM;
            const _Float16* Al_p = Pl + arow + (size_t)i0 * DIM;
            const _Float16* Bh_p = Qh + brow + (size_t)jt * 16 * DIM;
            const _Float16* Bl_p = Ql + brow + (size_t)jt * 16 * DIM;
#pragma unroll
            for (int h = 0; h < 2; ++h) {            // K = 64 in two K=32 steps
                const int d0 = h * 32 + dsel;
                v16h Ah = sig_load_frag(Ah_p + d0);
                v16h Al = sig_load_frag(Al_p + d0);
                v16h Bh = sig_load_frag(Bh_p + d0);
                v16h Bl = sig_load_frag(Bl_p + d0);
                acc = __builtin_amdgcn_wmma_f32_16x16x32_f16(false, Ah, false, Bh,
                        (short)0, acc, false, false);
                acc = __builtin_amdgcn_wmma_f32_16x16x32_f16(false, Ah, false, Bl,
                        (short)0, acc, false, false);
                acc = __builtin_amdgcn_wmma_f32_16x16x32_f16(false, Al, false, Bh,
                        (short)0, acc, false, false);
            }
            const int n  = jt * 16 + (lane & 15);    // C/D layout: N = lane&15
            const int mb = (lane >> 4) * 8;          // M = mb + r
#pragma unroll
            for (int r = 0; r < 8; ++r) incS[mb + r][n] = acc[r];
        }
        __syncthreads();

        // ---- Phase 2: each PDE row = elementwise term + 256-wide prefix sum
        const int rmax = (255 - i0 < 16) ? (255 - i0) : 16;
        for (int r = 0; r < rmax; ++r) {
            const float kj   = Krow[tid];
            const float kjm1 = (tid > 0) ? Krow[tid - 1] : 0.0f;
            float v = (tid == 0) ? 1.0f
                                 : (kj + kjm1 * (incS[r][tid - 1] - 1.0f));
            // wave32 inclusive scan
            float s = v;
#pragma unroll
            for (int off = 1; off < 32; off <<= 1) {
                float t = __shfl_up(s, off, 32);
                if (lane >= off) s += t;
            }
            if (lane == 31) waveSum[wave] = s;
            __syncthreads();                         // waveSum visible; Krow reads done
            float pre = 0.0f;
#pragma unroll
            for (int w = 0; w < 8; ++w) if (w < wave) pre += waveSum[w];
            s += pre;
            Krow[tid] = s;                           // K[i+1, tid]
            __syncthreads();                         // protect waveSum/Krow next iter
        }
        __syncthreads();                             // incS reuse safety
    }

    if (tid == 0) {
        const float w = (mat == 2) ? (-2.0f / 256.0f) : (1.0f / 256.0f);
        atomicAdd(out, w * Krow[255]);               // K[:, :, -1, -1], mean-weighted
    }
}

extern "C" void kernel_launch(void* const* d_in, const int* in_sizes, int n_in,
                              void* d_out, int out_size, void* d_ws, size_t ws_size,
                              hipStream_t stream) {
    const float* X = (const float*)d_in[0];
    const float* Y = (const float*)d_in[1];
    float* out = (float*)d_out;

    _Float16* ws  = (_Float16*)d_ws;     // 4 * 512 KB = 2 MB of scratch
    _Float16* dXh = ws;
    _Float16* dXl = dXh + NPTS;
    _Float16* dYh = dXl + NPTS;
    _Float16* dYl = dYh + NPTS;

    sig_zero_out<<<1, 1, 0, stream>>>(out);
    sig_prep<<<(NPTS + 255) / 256, 256, 0, stream>>>(X, Y, dXh, dXl, dYh, dYl, out);
    sig_pde<<<768, 256, 0, stream>>>(dXh, dXl, dYh, dYl, out);
}
